// AA2_Module_75359496175785
// MI455X (gfx1250) — matrix operand order repaired and checked by reading, not compiled, not run on hardware
//
#include <hip/hip_runtime.h>
#include <hip/hip_bf16.h>

// ---------------------------------------------------------------------------
// Types for CDNA5 WMMA (wave32): v16bf = A/B fragment, v8f = C/D accumulator
// ---------------------------------------------------------------------------
typedef __attribute__((ext_vector_type(16))) __bf16 v16bf;
typedef __attribute__((ext_vector_type(8)))  __bf16 bf16x8;
typedef __attribute__((ext_vector_type(8)))  float  v8f;

union FragU { v16bf v; bf16x8 h[2]; };

// A-matrix fragment (16x32 bf16, MxK) from row-major storage with leading dim ld.
// ISA layout: lanes 0-15 hold M=0..15, K = lo*8..lo*8+7 (VGPR0-3) and +16 (VGPR4-7),
// where lo = lane/16 selects K-subgroup.
__device__ __forceinline__ v16bf load_frag_a(const __bf16* base, int ld) {
  const int lane = threadIdx.x & 31;
  const int m = lane & 15, lo = lane >> 4;
  const __bf16* p = base + m * ld + lo * 8;
  FragU u;
  u.h[0] = *reinterpret_cast<const bf16x8*>(p);        // K = lo*8 .. +7
  u.h[1] = *reinterpret_cast<const bf16x8*>(p + 16);   // K = 16 + lo*8 .. +7
  return u.v;
}

// B-matrix fragment (32x16 bf16, KxN) from *column-major* storage, i.e. from a
// row-major (N x K) array Bs with leading dim ld: lane holds column n = lane%16,
// K = (lane/16)*8 .. +7 and 16 + (lane/16)*8 .. +7 (same as A).
__device__ __forceinline__ v16bf load_frag_b(const __bf16* base, int ld) {
  const int lane = threadIdx.x & 31;
  const int n = lane & 15, kg = lane >> 4;
  const __bf16* p = base + n * ld + kg * 8;
  FragU u;
  u.h[0] = *reinterpret_cast<const bf16x8*>(p);
  u.h[1] = *reinterpret_cast<const bf16x8*>(p + 16);
  return u.v;
}

__device__ __forceinline__ v8f wmma_bf16(v16bf a, v16bf b, v8f c) {
  return __builtin_amdgcn_wmma_f32_16x16x32_bf16(false, a, false, b, (short)0, c,
                                                 false, false);
}

// ---------------------------------------------------------------------------
// Kernel 1: adaptive 8x8 pool of x : (B*C) blocks, each pools one 128x128 plane
// into 64 block means. xp[bc][j], j = jh*8+jw.
// ---------------------------------------------------------------------------
__global__ void pool_kernel(const float* __restrict__ x, float* __restrict__ xp) {
  __shared__ float red[256];
  const int bc = blockIdx.x;
  const float* base = x + (size_t)bc * 16384;
  const int t = threadIdx.x;
  const int j = t & 63, part = t >> 6;
  const int jh = j >> 3, jw = j & 7;
  float s = 0.f;
  const int r0 = jh * 16 + part * 4;
#pragma unroll
  for (int rr = 0; rr < 4; ++rr) {
    const float* row = base + (size_t)(r0 + rr) * 128 + jw * 16;
#pragma unroll
    for (int cc = 0; cc < 16; ++cc) s += row[cc];
  }
  red[t] = s;
  __syncthreads();
  if (part == 0) {
    float tot = red[j] + red[j + 64] + red[j + 128] + red[j + 192];
    xp[(size_t)bc * 64 + j] = tot * (1.0f / 256.0f);
  }
}

// ---------------------------------------------------------------------------
// Kernel 2: per-batch key projections (tiny GEMMs). Writes, in bf16:
//   kt  = (Wk*xp + bk)^T       layout [b][j][c]  (64 x 128) -- for energy phase
//   kr  = gamma*(Wk*xp + bk)   layout [b][c][j]  (128 x 64) -- for output phase
// for both branches. gamma folded here so both branches share one accumulator.
// ---------------------------------------------------------------------------
__global__ void kproj_kernel(const float* __restrict__ xp,
                             const float* __restrict__ Wk,  const float* __restrict__ bk,
                             const float* __restrict__ gamma,
                             const float* __restrict__ Wk1, const float* __restrict__ bk1,
                             const float* __restrict__ gamma1,
                             __bf16* __restrict__ kt0, __bf16* __restrict__ kr0,
                             __bf16* __restrict__ kt1, __bf16* __restrict__ kr1) {
  __shared__ float xs[128 * 64];
  const int b = blockIdx.x;
  const int t = threadIdx.x;
  for (int i = t; i < 128 * 64; i += 256) xs[i] = xp[(size_t)b * 128 * 64 + i];
  __syncthreads();
  const float g0 = gamma[0], g1 = gamma1[0];
  for (int i = t; i < 128 * 64; i += 256) {
    const int c = i >> 6, j = i & 63;
    const float* wr0 = Wk + (size_t)c * 128;
    const float* wr1 = Wk1 + (size_t)c * 128;
    float d0 = bk[c], d1 = bk1[c];
#pragma unroll 8
    for (int ci = 0; ci < 128; ++ci) {
      const float xv = xs[ci * 64 + j];
      d0 = fmaf(wr0[ci], xv, d0);
      d1 = fmaf(wr1[ci], xv, d1);
    }
    kt0[((size_t)b * 64 + j) * 128 + c] = (__bf16)d0;
    kr0[((size_t)b * 128 + c) * 64 + j] = (__bf16)(g0 * d0);
    kt1[((size_t)b * 64 + j) * 128 + c] = (__bf16)d1;
    kr1[((size_t)b * 128 + c) * 64 + j] = (__bf16)(g1 * d1);
  }
}

// ---------------------------------------------------------------------------
// Kernel 3: fused attention. Grid = B * (N/128) workgroups, 256 threads (8 waves).
// LDS (dynamic, 120 KB): sWq 128x136, sXt 128x136 (x^T tile), sQt 128x136 (Q^T),
// sAtt 128x72 (attn). Padded leading dims keep 16B alignment + spread banks.
// ---------------------------------------------------------------------------
#define LDW 136
#define LDA 72
#define SMEM_BYTES ((3 * 128 * LDW + 128 * LDA) * 2)

__global__ void __launch_bounds__(256)
attn_main(const float* __restrict__ x, const float* __restrict__ Wq,
          const float* __restrict__ bq,
          const float* __restrict__ alpha0p, const float* __restrict__ alpha1p,
          const __bf16* __restrict__ kt0, const __bf16* __restrict__ kr0,
          const __bf16* __restrict__ kt1, const __bf16* __restrict__ kr1,
          float* __restrict__ out) {
  extern __shared__ char smem[];
  __bf16* sWq  = (__bf16*)smem;              // [o][ci]   128 x LDW
  __bf16* sXt  = sWq + 128 * LDW;            // [n][ci]   128 x LDW
  __bf16* sQt  = sXt + 128 * LDW;            // [n][co]   128 x LDW
  __bf16* sAtt = sQt + 128 * LDW;            // [n][j]    128 x LDA

  const int b   = blockIdx.x >> 7;
  const int n0  = (blockIdx.x & 127) << 7;
  const int t   = threadIdx.x;
  const int w   = t >> 5;                    // wave id 0..7
  const int lane = t & 31;
  const int ln16 = lane & 15, hl = lane >> 4;

  // ---- stage Wq and x tile (f32 -> bf16, x transposed) ----
  for (int i = t; i < 128 * 128; i += 256) {
    const int o = i >> 7, ci = i & 127;
    sWq[o * LDW + ci] = (__bf16)Wq[i];
  }
  for (int i = t; i < 128 * 128; i += 256) {
    const int ci = i >> 7, n = i & 127;      // consecutive threads -> consecutive n
    sXt[n * LDW + ci] = (__bf16)x[((size_t)(b * 128 + ci)) * 16384 + n0 + n];
  }
  __syncthreads();

  // ---- phase A: Q = Wq @ X  (each wave owns 16 output channels) ----
  {
    const int co0 = w * 16;
    v16bf afr[4];
#pragma unroll
    for (int kc = 0; kc < 4; ++kc)
      afr[kc] = load_frag_a(sWq + co0 * LDW + kc * 32, LDW);
    float bias[8];
#pragma unroll
    for (int r = 0; r < 8; ++r) bias[r] = bq[co0 + 8 * hl + r];
#pragma unroll
    for (int nt = 0; nt < 8; ++nt) {
      v8f acc = {0.f, 0.f, 0.f, 0.f, 0.f, 0.f, 0.f, 0.f};
#pragma unroll
      for (int kc = 0; kc < 4; ++kc) {
        v16bf bfr = load_frag_b(sXt + (nt * 16) * LDW + kc * 32, LDW);
        acc = wmma_bf16(afr[kc], bfr, acc);
      }
      const int n = nt * 16 + ln16;
#pragma unroll
      for (int r = 0; r < 8; ++r)
        sQt[n * LDW + co0 + 8 * hl + r] = (__bf16)(acc[r] + bias[r]);
    }
  }
  __syncthreads();

  v8f oacc[8];
#pragma unroll
  for (int tt = 0; tt < 8; ++tt)
    oacc[tt] = (v8f){0.f, 0.f, 0.f, 0.f, 0.f, 0.f, 0.f, 0.f};

  const float alph0 = alpha0p[0], alph1 = alpha1p[0];
  const __bf16* ktb0 = kt0 + (size_t)b * 64 * 128;
  const __bf16* ktb1 = kt1 + (size_t)b * 64 * 128;
  const __bf16* krb0 = kr0 + (size_t)b * 128 * 64;
  const __bf16* krb1 = kr1 + (size_t)b * 128 * 64;

#pragma unroll
  for (int br = 0; br < 2; ++br) {
    const __bf16* kt = br ? ktb1 : ktb0;
    const __bf16* kr = br ? krb1 : krb0;
    const float alpha = br ? alph1 : alph0;

    // ---- phase B: energy^T = K^T(64xC) @ Q(CxN); wave owns n-cols [16w,16w+16) ----
    v8f e[4];
#pragma unroll
    for (int jt = 0; jt < 4; ++jt)
      e[jt] = (v8f){0.f, 0.f, 0.f, 0.f, 0.f, 0.f, 0.f, 0.f};
    v16bf bfr[4];
#pragma unroll
    for (int kc = 0; kc < 4; ++kc)
      bfr[kc] = load_frag_b(sQt + (w * 16) * LDW + kc * 32, LDW);
#pragma unroll
    for (int jt = 0; jt < 4; ++jt) {
#pragma unroll
      for (int kc = 0; kc < 4; ++kc) {
        v16bf afr = load_frag_a(kt + (jt * 16) * 128 + kc * 32, 128);
        e[jt] = wmma_bf16(afr, bfr[kc], e[jt]);
      }
    }

    // ---- softmax over the 64 keys for each position n (per-lane + xor16) ----
    float mx = -3.4e38f;
#pragma unroll
    for (int jt = 0; jt < 4; ++jt)
#pragma unroll
      for (int r = 0; r < 8; ++r) {
        e[jt][r] *= alpha;
        mx = fmaxf(mx, e[jt][r]);
      }
    mx = fmaxf(mx, __shfl_xor(mx, 16, 32));
    float sum = 0.f;
#pragma unroll
    for (int jt = 0; jt < 4; ++jt)
#pragma unroll
      for (int r = 0; r < 8; ++r) {
        const float p = __expf(e[jt][r] - mx);
        e[jt][r] = p;
        sum += p;
      }
    sum += __shfl_xor(sum, 16, 32);
    const float inv = 1.0f / sum;
    const int n = w * 16 + ln16;
#pragma unroll
    for (int jt = 0; jt < 4; ++jt)
#pragma unroll
      for (int r = 0; r < 8; ++r)
        sAtt[n * LDA + jt * 16 + 8 * hl + r] = (__bf16)(e[jt][r] * inv);
    __syncthreads();

    // ---- phase C: out(CxN) += (gamma*K)(Cx64) @ attn^T(64xN) ----
    {
      const int c0 = w * 16;
      v16bf afr2[2];
#pragma unroll
      for (int s = 0; s < 2; ++s)
        afr2[s] = load_frag_a(kr + c0 * 64 + s * 32, 64);
#pragma unroll
      for (int tt = 0; tt < 8; ++tt) {
#pragma unroll
        for (int s = 0; s < 2; ++s) {
          v16bf bfr2 = load_frag_b(sAtt + (tt * 16) * LDA + s * 32, LDA);
          oacc[tt] = wmma_bf16(afr2[s], bfr2, oacc[tt]);
        }
      }
    }
    __syncthreads();   // protect sAtt before next branch rewrites it
  }

  // ---- epilogue: + x (residual), coalesced along n within each half-wave ----
  {
    const int c0 = w * 16;
#pragma unroll
    for (int tt = 0; tt < 8; ++tt) {
      const int n = n0 + tt * 16 + ln16;
#pragma unroll
      for (int r = 0; r < 8; ++r) {
        const size_t idx = ((size_t)(b * 128 + c0 + 8 * hl + r)) * 16384 + n;
        out[idx] = oacc[tt][r] + x[idx];
      }
    }
  }
}

// ---------------------------------------------------------------------------
// Launch
// ---------------------------------------------------------------------------
extern "C" void kernel_launch(void* const* d_in, const int* in_sizes, int n_in,
                              void* d_out, int out_size, void* d_ws, size_t ws_size,
                              hipStream_t stream) {
  (void)in_sizes; (void)n_in; (void)out_size; (void)ws_size;
  const float* x      = (const float*)d_in[0];
  const float* Wq     = (const float*)d_in[1];
  const float* bq     = (const float*)d_in[2];
  const float* Wk     = (const float*)d_in[3];
  const float* bk     = (const float*)d_in[4];
  const float* Wk1    = (const float*)d_in[5];
  const float* bk1    = (const float*)d_in[6];
  const float* gamma  = (const float*)d_in[7];
  const float* gamma1 = (const float*)d_in[8];
  const float* aphal  = (const float*)d_in[9];
  const float* aphal1 = (const float*)d_in[10];
  float* out = (float*)d_out;

  // workspace carve-up (all 128B-aligned offsets)
  char* ws = (char*)d_ws;
  float*  xp  = (float*)ws;                                   // 8*128*64*4   = 256 KB
  __bf16* kt0 = (__bf16*)(ws + 262144);                       // 8*64*128*2   = 128 KB
  __bf16* kr0 = (__bf16*)(ws + 262144 + 1 * 131072);
  __bf16* kt1 = (__bf16*)(ws + 262144 + 2 * 131072);
  __bf16* kr1 = (__bf16*)(ws + 262144 + 3 * 131072);

  pool_kernel<<<8 * 128, 256, 0, stream>>>(x, xp);
  kproj_kernel<<<8, 256, 0, stream>>>(xp, Wk, bk, gamma, Wk1, bk1, gamma1,
                                      kt0, kr0, kt1, kr1);

  (void)hipFuncSetAttribute((const void*)attn_main,
                            hipFuncAttributeMaxDynamicSharedMemorySize, SMEM_BYTES);
  attn_main<<<8 * 128, 256, SMEM_BYTES, stream>>>(x, Wq, bq, aphal, aphal1,
                                                  kt0, kr0, kt1, kr1, out);
}
